// GINEModule_31164282699784
// MI455X (gfx1250) — matrix operand order repaired and checked by reading, not compiled, not run on hardware
//
#include <hip/hip_runtime.h>

// ---------------------------------------------------------------------------
// GINE 3-layer GNN forward for MI455X (gfx1250), bf16 WMMA f32-accumulate.
//
// Input order (setup_inputs dict insertion order, params flattened in
// insertion order): 0:x 1:edge_index(i64) 2:edge_attr,
// then per layer l (base 3+10*l): We,be,W1,b1,W2,b2,eps,gamma,beta,a
// then 33:Wout 34:bout.
// ---------------------------------------------------------------------------

typedef __attribute__((ext_vector_type(16))) __bf16           v16bf;
typedef __attribute__((ext_vector_type(8)))  float            v8f;

#define NNODES   50000
#define NEDGES   800000
#define HIDF     128
#define EDGEF    64
#define OUTF     64
#define NODE_TILES (NNODES / 16)   // 3125
#define EDGE_TILES (NEDGES / 16)   // 50000

// native hardware conversion (v_cvt_pk_bf16_f32), RNE
static __device__ __forceinline__ __bf16 f2bf(float f) { return (__bf16)f; }
static __device__ __forceinline__ unsigned short f2bf_bits(float f) {
  return __builtin_bit_cast(unsigned short, (__bf16)f);
}

static __device__ __forceinline__ v8f wmma_bf16(v16bf a, v16bf b, v8f c) {
  return __builtin_amdgcn_wmma_f32_16x16x32_bf16(
      false, a, false, b, (short)0, c, false, false);
}

// A-matrix 16x32 bf16 fragment (ISA 7.12.2): lane L holds row M=L&15;
// halves 0..7 = K kbase+koff+e, halves 8..15 = K kbase+16+koff+e, koff=(L>>4)*8
static __device__ __forceinline__ v16bf a_frag_f32(const float* __restrict__ rowp,
                                                   int kbase, int lane) {
  int koff = (lane >> 4) * 8;
  v16bf u;
#pragma unroll
  for (int e = 0; e < 8; ++e) u[e] = f2bf(rowp[kbase + koff + e]);
#pragma unroll
  for (int e = 0; e < 8; ++e) u[8 + e] = f2bf(rowp[kbase + 16 + koff + e]);
  return u;
}

// Load a pre-packed B fragment: [frag][lane][16 bf16], 32B/lane, 32B aligned.
static __device__ __forceinline__ v16bf load_bfrag(const unsigned short* __restrict__ base,
                                                   int frag, int lane) {
  return *(const v16bf*)(base + ((unsigned)frag * 32u + (unsigned)lane) * 16u);
}

// ---------------------------------------------------------------------------
// Weight prep: pack W[K x N] (row-major, leading dim ldb==N) into WMMA
// B-fragment layout (lane = col, half e<8: K (L>>4)*8+e, e>=8: K 16+(L>>4)*8+e-8).
// grid = (K/32)*(N/16) blocks of 32 threads; frag = kf*ntiles + nt.
// ---------------------------------------------------------------------------
__global__ void prep_wfrag(const float* __restrict__ W, int ldb, int ntiles,
                           unsigned short* __restrict__ dst) {
  int frag = blockIdx.x;
  int lane = threadIdx.x & 31;
  int kf = frag / ntiles, nt = frag % ntiles;
  int col  = nt * 16 + (lane & 15);
  int kadd = (lane >> 4) * 8;
  int kbase = kf * 32;
  v16bf u;
#pragma unroll
  for (int e = 0; e < 16; ++e)
    u[e] = f2bf(W[(unsigned)(kbase + kadd + e + (e >= 8 ? 8 : 0)) * (unsigned)ldb + (unsigned)col]);
  *(v16bf*)(dst + ((unsigned)frag * 32u + (unsigned)lane) * 16u) = u;
}

__global__ void zero4(float4* __restrict__ p, int n4) {
  int i = blockIdx.x * blockDim.x + threadIdx.x;
  if (i < n4) p[i] = make_float4(0.f, 0.f, 0.f, 0.f);
}

// ---------------------------------------------------------------------------
// Edge kernel: one wave per 16-edge tile.
// e = edge_attr @ We (WMMA) ; msg = relu(x[src] + e + be) ; atomic agg[dst]+=
// eidx32: low dwords of the little-endian int64 edge_index (ids < 50000).
// ---------------------------------------------------------------------------
__global__ __launch_bounds__(256) void gine_edge(
    const float* __restrict__ x, const float* __restrict__ eattr,
    const int* __restrict__ eidx32, const unsigned short* __restrict__ weF,
    const float* __restrict__ be, float* __restrict__ agg, int numTiles) {
  int lane = threadIdx.x & 31;
  int tile = blockIdx.x * 8 + (threadIdx.x >> 5);
  if (tile >= numTiles) return;  // wave-uniform
  int ebase = tile * 16;

  int mrow = (lane >> 4) * 8;
  unsigned srcs[8], dsts[8];
#pragma unroll
  for (int r = 0; r < 8; ++r) {
    unsigned e = (unsigned)(ebase + mrow + r);
    srcs[r] = (unsigned)eidx32[2u * e];                     // low dword of i64
    dsts[r] = (unsigned)eidx32[2u * ((unsigned)NEDGES + e)];
  }
  const float* arow = eattr + (size_t)(ebase + (lane & 15)) * EDGEF;
  // stream-prefetch the next tile's edge_attr rows (global_prefetch_b8)
  __builtin_prefetch(arow + 16 * EDGEF, 0, 0);
  v16bf a0 = a_frag_f32(arow, 0, lane);
  v16bf a1 = a_frag_f32(arow, 32, lane);

  int ncol = lane & 15;
#pragma unroll
  for (int nt = 0; nt < 8; ++nt) {
    v8f c = {0.f, 0.f, 0.f, 0.f, 0.f, 0.f, 0.f, 0.f};
    c = wmma_bf16(a0, load_bfrag(weF, 0 * 8 + nt, lane), c);
    c = wmma_bf16(a1, load_bfrag(weF, 1 * 8 + nt, lane), c);
    unsigned n = (unsigned)(nt * 16 + ncol);
    float bev = be[n];
#pragma unroll
    for (int r = 0; r < 8; ++r) {
      // 32-bit element offsets (max 25.6MB byte offset) -> SADDR+voffset form
      float m = c[r] + bev + x[srcs[r] * (unsigned)HIDF + n];
      m = m > 0.f ? m : 0.f;
      atomicAdd(&agg[dsts[r] * (unsigned)HIDF + n], m);
    }
  }
}

// ---------------------------------------------------------------------------
// Node kernel: one wave per 16-node tile.
// h = (1+eps)*x + agg ; h1 = relu(h@W1+b1) (LDS transpose) ; h2 = h1@W2+b2
// plus BN sum/sumsq atomics into stats[0:128]/[128:256].
// ---------------------------------------------------------------------------
__global__ __launch_bounds__(256) void gine_node(
    const float* __restrict__ x, const float* __restrict__ agg,
    const float* __restrict__ epsp,
    const unsigned short* __restrict__ w1F, const float* __restrict__ b1,
    const unsigned short* __restrict__ w2F, const float* __restrict__ b2,
    float* __restrict__ h2, float* __restrict__ stats, int numTiles) {
  __shared__ unsigned short lds[8][16 * HIDF];  // 32KB: per-wave 16x128 bf16
  int lane = threadIdx.x & 31;
  int wid  = threadIdx.x >> 5;
  int tile = blockIdx.x * 8 + wid;
  bool active = tile < numTiles;
  int t = active ? tile : 0;  // clamp so all waves run full EXEC through WMMA

  float eps1 = 1.0f + epsp[0];
  unsigned rbase = ((unsigned)t * 16u + (unsigned)(lane & 15)) * (unsigned)HIDF;
  const float* xr = x + rbase;
  const float* gr = agg + rbase;

  int koff = (lane >> 4) * 8;
  v16bf ha[4];
#pragma unroll
  for (int kf = 0; kf < 4; ++kf) {
    int kb = kf * 32;
    v16bf u;
#pragma unroll
    for (int e = 0; e < 8; ++e) {
      int k = kb + koff + e;
      u[e] = f2bf(eps1 * xr[k] + gr[k]);
    }
#pragma unroll
    for (int e = 0; e < 8; ++e) {
      int k = kb + 16 + koff + e;
      u[8 + e] = f2bf(eps1 * xr[k] + gr[k]);
    }
    ha[kf] = u;
  }

  unsigned short* myLds = lds[wid];
  int ncol = lane & 15;
  int mbase = (lane >> 4) * 8;
#pragma unroll
  for (int nt = 0; nt < 8; ++nt) {
    v8f c = {0.f, 0.f, 0.f, 0.f, 0.f, 0.f, 0.f, 0.f};
#pragma unroll
    for (int kf = 0; kf < 4; ++kf)
      c = wmma_bf16(ha[kf], load_bfrag(w1F, kf * 8 + nt, lane), c);
    float b1v = b1[nt * 16 + ncol];
#pragma unroll
    for (int r = 0; r < 8; ++r) {
      float v = c[r] + b1v;
      v = v > 0.f ? v : 0.f;
      myLds[(mbase + r) * HIDF + nt * 16 + ncol] = f2bf_bits(v);
    }
  }
  __syncthreads();

  // rebuild A fragments of h1 from LDS (bf16, already in target precision)
  const unsigned short* lrow = myLds + (lane & 15) * HIDF;
  v16bf g[4];
#pragma unroll
  for (int kf = 0; kf < 4; ++kf) {
    int kb = kf * 32;
    v16bf u;
#pragma unroll
    for (int e = 0; e < 8; ++e)
      u[e] = __builtin_bit_cast(__bf16, lrow[kb + koff + e]);
#pragma unroll
    for (int e = 0; e < 8; ++e)
      u[8 + e] = __builtin_bit_cast(__bf16, lrow[kb + 16 + koff + e]);
    g[kf] = u;
  }

#pragma unroll
  for (int nt = 0; nt < 8; ++nt) {
    v8f c = {0.f, 0.f, 0.f, 0.f, 0.f, 0.f, 0.f, 0.f};
#pragma unroll
    for (int kf = 0; kf < 4; ++kf)
      c = wmma_bf16(g[kf], load_bfrag(w2F, kf * 8 + nt, lane), c);
    unsigned n = (unsigned)(nt * 16 + ncol);
    float b2v = b2[n];
    float s = 0.f, sq = 0.f;
    float hv[8];
#pragma unroll
    for (int r = 0; r < 8; ++r) {
      float v = c[r] + b2v;
      hv[r] = v;
      s += v;
      sq += v * v;
    }
    // lanes L and L^16 hold the same output column: pair-reduce, halve atomics
    s  += __shfl_xor(s, 16, 32);
    sq += __shfl_xor(sq, 16, 32);
    if (active) {
#pragma unroll
      for (int r = 0; r < 8; ++r)
        h2[((unsigned)t * 16u + (unsigned)(mbase + r)) * (unsigned)HIDF + n] = hv[r];
      if (lane < 16) {
        atomicAdd(&stats[n], s);
        atomicAdd(&stats[HIDF + n], sq);
      }
    }
  }
}

// ---------------------------------------------------------------------------
// BatchNorm (batch stats) + PReLU, elementwise; writes next-layer x.
// ---------------------------------------------------------------------------
__global__ void bn_prelu(const float* __restrict__ h2, const float* __restrict__ stats,
                         const float* __restrict__ gamma, const float* __restrict__ beta,
                         const float* __restrict__ slope, float* __restrict__ xout,
                         int total, float invN) {
  int i = blockIdx.x * blockDim.x + threadIdx.x;
  if (i >= total) return;
  int col = i & (HIDF - 1);
  float mu  = stats[col] * invN;
  float ex2 = stats[HIDF + col] * invN;
  float var = ex2 - mu * mu;
  float v = (h2[i] - mu) * rsqrtf(var + 1e-5f) * gamma[col] + beta[col];
  xout[i] = v > 0.f ? v : slope[col] * v;
}

// ---------------------------------------------------------------------------
// Output GEMM: out = x @ Wout + bout ([50000,128] @ [128,64])
// ---------------------------------------------------------------------------
__global__ __launch_bounds__(256) void out_gemm(
    const float* __restrict__ x, const unsigned short* __restrict__ woF,
    const float* __restrict__ bout, float* __restrict__ out, int numTiles) {
  int lane = threadIdx.x & 31;
  int tile = blockIdx.x * 8 + (threadIdx.x >> 5);
  bool active = tile < numTiles;
  int t = active ? tile : 0;

  const float* xr = x + ((unsigned)t * 16u + (unsigned)(lane & 15)) * (unsigned)HIDF;
  v16bf a[4];
#pragma unroll
  for (int kf = 0; kf < 4; ++kf) a[kf] = a_frag_f32(xr, kf * 32, lane);

  int ncol = lane & 15;
  int mbase = (lane >> 4) * 8;
#pragma unroll
  for (int nt = 0; nt < 4; ++nt) {
    v8f c = {0.f, 0.f, 0.f, 0.f, 0.f, 0.f, 0.f, 0.f};
#pragma unroll
    for (int kf = 0; kf < 4; ++kf)
      c = wmma_bf16(a[kf], load_bfrag(woF, kf * 4 + nt, lane), c);
    unsigned n = (unsigned)(nt * 16 + ncol);
    float bv = bout[n];
    if (active) {
#pragma unroll
      for (int r = 0; r < 8; ++r)
        out[((unsigned)t * 16u + (unsigned)(mbase + r)) * (unsigned)OUTF + n] = c[r] + bv;
    }
  }
}

// ---------------------------------------------------------------------------
extern "C" void kernel_launch(void* const* d_in, const int* in_sizes, int n_in,
                              void* d_out, int out_size, void* d_ws, size_t ws_size,
                              hipStream_t stream) {
  (void)in_sizes; (void)n_in; (void)out_size; (void)ws_size;
  const float* x0    = (const float*)d_in[0];
  const int*   eidx32= (const int*)d_in[1];   // int64 indices; read low dwords
  const float* eattr = (const float*)d_in[2];
  const float* Wout  = (const float*)d_in[33];
  const float* bout  = (const float*)d_in[34];

  // workspace layout (floats)
  float* ws    = (float*)d_ws;
  float* xcur  = ws;                        // 6.4M
  float* agg   = ws + 6400000;              // 6.4M
  float* h2    = ws + 12800000;             // 6.4M
  float* stats = ws + 19200000;             // 256
  unsigned short* fragb = (unsigned short*)(ws + 19200256);  // 32B aligned
  // frag area (ushort units): per layer 80 frags * 512 = 40960; layout:
  // We@+0 (16 frags), W1@+8192 (32), W2@+24576 (32); Wout @ 122880 (16)

  // --- pack all weights into WMMA B-fragment layout (bf16) ---
  for (int l = 0; l < 3; ++l) {
    const float* We = (const float*)d_in[3 + l * 10 + 0];
    const float* W1 = (const float*)d_in[3 + l * 10 + 2];
    const float* W2 = (const float*)d_in[3 + l * 10 + 4];
    unsigned short* lb = fragb + (size_t)l * 40960;
    prep_wfrag<<<16, 32, 0, stream>>>(We, 128, 8, lb);           // 64x128
    prep_wfrag<<<32, 32, 0, stream>>>(W1, 128, 8, lb + 8192);    // 128x128
    prep_wfrag<<<32, 32, 0, stream>>>(W2, 128, 8, lb + 24576);   // 128x128
  }
  prep_wfrag<<<16, 32, 0, stream>>>(Wout, 64, 4, fragb + 122880); // 128x64

  for (int l = 0; l < 3; ++l) {
    const float* be    = (const float*)d_in[3 + l * 10 + 1];
    const float* b1    = (const float*)d_in[3 + l * 10 + 3];
    const float* b2    = (const float*)d_in[3 + l * 10 + 5];
    const float* eps   = (const float*)d_in[3 + l * 10 + 6];
    const float* gamma = (const float*)d_in[3 + l * 10 + 7];
    const float* beta  = (const float*)d_in[3 + l * 10 + 8];
    const float* slope = (const float*)d_in[3 + l * 10 + 9];
    unsigned short* lb = fragb + (size_t)l * 40960;
    const float* xin = (l == 0) ? x0 : xcur;

    zero4<<<6250, 256, 0, stream>>>((float4*)agg, 1600000);
    zero4<<<1, 256, 0, stream>>>((float4*)stats, 64);

    gine_edge<<<EDGE_TILES / 8, 256, 0, stream>>>(
        xin, eattr, eidx32, lb, be, agg, EDGE_TILES);

    gine_node<<<(NODE_TILES + 7) / 8, 256, 0, stream>>>(
        xin, agg, eps, lb + 8192, b1, lb + 24576, b2, h2, stats, NODE_TILES);

    bn_prelu<<<(NNODES * HIDF) / 256, 256, 0, stream>>>(
        h2, stats, gamma, beta, slope, xcur, NNODES * HIDF, 1.0f / (float)NNODES);
  }

  out_gemm<<<(NODE_TILES + 7) / 8, 256, 0, stream>>>(
      xcur, fragb + 122880, bout, (float*)d_out, NODE_TILES);
}
